// VectorQuantizer2_13692355739657
// MI455X (gfx1250) — matrix operand order repaired and checked by reading, not compiled, not run on hardware
//
#include <hip/hip_runtime.h>
#include <hip/hip_bf16.h>

typedef __attribute__((ext_vector_type(16))) _Float16 v16h;
typedef __attribute__((ext_vector_type(8)))  float    v8f;

#define ELEMS  (8*32*64*64)   // 1048576
#define VOCAB  4096
#define CVAE   32

// ---------------------------------------------------------------- init
__global__ __launch_bounds__(256) void init_kernel(const float* __restrict__ f,
                                                   float* __restrict__ f_hat,
                                                   float* __restrict__ f_rest,
                                                   float* __restrict__ acc) {
    int i = blockIdx.x * 256 + threadIdx.x;
    if (i < ELEMS) { f_hat[i] = 0.0f; f_rest[i] = f[i]; }
    if (i < 2) acc[i] = 0.0f;
}

// ------------------------------------------------- normalized f16 codebook
__global__ __launch_bounds__(256) void cbn_kernel(const float* __restrict__ cb,
                                                  _Float16* __restrict__ cbn) {
    int r = blockIdx.x * 256 + threadIdx.x;
    if (r >= VOCAB) return;
    const float* row = cb + r * CVAE;
    float s = 0.0f;
    #pragma unroll
    for (int c = 0; c < CVAE; ++c) s += row[c] * row[c];
    float inv = rsqrtf(fmaxf(s, 1e-24f));
    #pragma unroll
    for (int c = 0; c < CVAE; ++c) cbn[r * CVAE + c] = (_Float16)(row[c] * inv);
}

// ------------------------------------------- area downsample -> [N][C] rows
__global__ __launch_bounds__(256) void downsample_nc_kernel(const float* __restrict__ fr,
                                                            float* __restrict__ rest_nc,
                                                            int pn) {
    int t = blockIdx.x * 256 + threadIdx.x;
    int N = 8 * pn * pn;
    if (t >= N * CVAE) return;
    int c = t & 31;
    int n = t >> 5;
    int q = n % pn, p = (n / pn) % pn, b = n / (pn * pn);
    int k = 64 / pn;
    const float* base = fr + (((b * 32 + c) * 64 + p * k) * 64 + q * k);
    float s = 0.0f;
    for (int i = 0; i < k; ++i)
        for (int j = 0; j < k; ++j)
            s += base[i * 64 + j];
    rest_nc[t] = s / (float)(k * k);
}

// ------------------------------------- WMMA cosine-sim argmax
// 256-thread block = 8 waves; block stages the full 256KB f16 codebook into
// LDS with gfx1250 async global->LDS copies (ASYNCcnt), then each wave runs
// a 16-row group: one v_wmma_f32_16x16x32_f16 per 16x16 tile (K=32 == CVAE),
// B-fragments read from LDS.
__global__ __launch_bounds__(256) void sim_argmax_kernel(const float* __restrict__ rest_nc,
                                                         const _Float16* __restrict__ cbn,
                                                         int* __restrict__ idx_out, int N) {
    extern __shared__ _Float16 smem[];   // [VOCAB*CVAE] codebook + reduction area
    const int tid  = threadIdx.x;
    const int lane = tid & 31;
    const int wave = tid >> 5;
    const int g    = lane >> 4;          // lane-half selector
    const int m    = lane & 15;
    const int rowbase = blockIdx.x * 128 + wave * 16;

    // ---- async-stage codebook: global -> LDS, tracked by ASYNCcnt ----
    {
        const uint32_t lds_base = (uint32_t)(uintptr_t)(void*)smem;  // flat lo32 == LDS offset
        for (int i = tid; i < (VOCAB * CVAE) / 8; i += 256) {        // 16384 x 16B chunks
            uint32_t lo = lds_base + (uint32_t)i * 16u;
            uint64_t ga = (uint64_t)(uintptr_t)(const void*)(cbn + i * 8);
            asm volatile("global_load_async_to_lds_b128 %0, %1, off"
                         :: "v"(lo), "v"(ga) : "memory");
        }
        asm volatile("s_wait_asynccnt 0x0" ::: "memory");
    }
    __syncthreads();

    // A fragment (16x32 f16), contiguous K order matching B:
    //   element h: K = 16g + h
    v16h a;
    {
        int row = rowbase + m;
        row = (row < N) ? row : (N - 1);          // clamp padded rows
        const float* rp = rest_nc + row * CVAE;
        #pragma unroll
        for (int h = 0; h < 16; ++h) {
            int v = h >> 1, p = h & 1;
            int k = 16 * g + 2 * v + p;
            a[h] = (_Float16)rp[k];
        }
    }

    float best[8];
    int   bidx[8];
    #pragma unroll
    for (int r = 0; r < 8; ++r) { best[r] = -3.4e38f; bidx[r] = 0; }

    for (int t = 0; t < VOCAB / 16; ++t) {
        const int col = t * 16 + m;
        // B fragment (32x16 f16): lane-half g holds K=16g..16g+15 of column `col`
        // -> one contiguous 32B LDS read (ds_load_b128 x2).
        v16h b = *(const v16h*)(smem + col * CVAE + 16 * g);
        v8f c = {};
        c = __builtin_amdgcn_wmma_f32_16x16x32_f16(false, a, false, b,
                                                   (short)0, c, false, false);
        #pragma unroll
        for (int r = 0; r < 8; ++r) {
            float v = c[r];                       // row = rowbase + r + 8g, this col
            if (v > best[r]) { best[r] = v; bidx[r] = col; }
        }
    }

    // deterministic cross-lane argmax (lowest index wins ties, jnp semantics)
    float* s_val = (float*)(smem + VOCAB * CVAE);  // 256*8 floats
    int*   s_idx = (int*)(s_val + 256 * 8);        // 256*8 ints
    #pragma unroll
    for (int r = 0; r < 8; ++r) { s_val[tid * 8 + r] = best[r]; s_idx[tid * 8 + r] = bidx[r]; }
    __syncthreads();
    if (lane < 16) {
        int rr = lane;                   // matrix row M within this wave's group
        int r = rr & 7, gg = rr >> 3;
        float bv = -3.4e38f; int bi = 0x7fffffff;
        for (int l = 0; l < 16; ++l) {
            int slot = (wave * 32 + gg * 16 + l) * 8 + r;
            float v = s_val[slot]; int i = s_idx[slot];
            if (v > bv || (v == bv && i < bi)) { bv = v; bi = i; }
        }
        int row = rowbase + rr;
        if (row < N) idx_out[row] = bi;
    }
}

// ---------------------------------------------- bicubic weight (PyTorch A=-0.75)
__device__ __forceinline__ float cubicw(float t) {
    const float A = -0.75f;
    t = fabsf(t);
    if (t <= 1.0f) return ((A + 2.0f) * t - (A + 3.0f)) * t * t + 1.0f;
    if (t < 2.0f)  return (((t - 5.0f) * t + 8.0f) * t - 4.0f) * A;
    return 0.0f;
}

// ------------------------------------- gather codes + bicubic upsample to 64x64
__global__ __launch_bounds__(256) void gather_upsample_kernel(const int* __restrict__ idx,
                                                              const float* __restrict__ cb,
                                                              float* __restrict__ h, int pn) {
    int n = blockIdx.x * 256 + threadIdx.x;
    if (n >= ELEMS) return;
    int x = n & 63, y = (n >> 6) & 63, c = (n >> 12) & 31, b = n >> 17;
    if (pn == 64) {
        h[n] = cb[idx[(b * 64 + y) * 64 + x] * CVAE + c];
        return;
    }
    float scale = (float)pn / 64.0f;
    float fy = (y + 0.5f) * scale - 0.5f;
    float fx = (x + 0.5f) * scale - 0.5f;
    int y0 = (int)floorf(fy), x0 = (int)floorf(fx);
    float acc = 0.0f;
    #pragma unroll
    for (int jy = 0; jy < 4; ++jy) {
        int py = y0 - 1 + jy;
        float wy = cubicw(fy - (float)py);
        int pyc = min(max(py, 0), pn - 1);
        #pragma unroll
        for (int jx = 0; jx < 4; ++jx) {
            int px = x0 - 1 + jx;
            float wx = cubicw(fx - (float)px);
            int pxc = min(max(px, 0), pn - 1);
            int code = idx[(b * pn + pyc) * pn + pxc];
            acc += wy * wx * cb[code * CVAE + c];
        }
    }
    h[n] = acc;
}

// --------------------------------- Phi: h*(1-r) + (conv3x3(h)+bias)*r, r=0.5
__global__ __launch_bounds__(256) void phi_conv_kernel(const float* __restrict__ hin,
                                                       const float* __restrict__ w,
                                                       const float* __restrict__ bias,
                                                       float* __restrict__ hout) {
    __shared__ float wl[CVAE * 9];
    int n = blockIdx.x * 256 + threadIdx.x;
    int x = n & 63, y = (n >> 6) & 63, co = (n >> 12) & 31, b = n >> 17;
    // whole block shares co (4096-elem plane, 256 | 4096)
    for (int i = threadIdx.x; i < CVAE * 9; i += 256) wl[i] = w[co * CVAE * 9 + i];
    __syncthreads();
    float acc = 0.0f;
    const float* in_b = hin + b * CVAE * 4096;
    for (int ci = 0; ci < CVAE; ++ci) {
        const float* plane = in_b + ci * 4096;
        const float* wc = wl + ci * 9;
        #pragma unroll
        for (int dy = 0; dy < 3; ++dy) {
            int yy = y + dy - 1;
            if (yy < 0 || yy > 63) continue;
            #pragma unroll
            for (int dx = 0; dx < 3; ++dx) {
                int xx = x + dx - 1;
                if (xx < 0 || xx > 63) continue;
                acc += plane[yy * 64 + xx] * wc[dy * 3 + dx];
            }
        }
    }
    hout[n] = hin[n] * 0.5f + (acc + bias[co]) * 0.5f;
}

// ------------------------- f_hat += h ; f_rest -= h ; per-block SSE partials
__global__ __launch_bounds__(256) void update_loss_kernel(const float* __restrict__ f,
                                                          float* __restrict__ f_hat,
                                                          float* __restrict__ f_rest,
                                                          const float* __restrict__ h,
                                                          float* __restrict__ partials) {
    const int tid = threadIdx.x, bid = blockIdx.x;   // 1024 blocks x 1024 elems
    float local = 0.0f;
    #pragma unroll
    for (int it = 0; it < 4; ++it) {
        int i = (bid * 4 + it) * 256 + tid;
        float hv = h[i];
        float fh = f_hat[i] + hv;
        f_hat[i]  = fh;
        f_rest[i] = f_rest[i] - hv;
        float d = fh - f[i];
        local += d * d;
    }
    __shared__ float red[256];
    red[tid] = local;
    __syncthreads();
    for (int s = 128; s > 0; s >>= 1) {
        if (tid < s) red[tid] += red[tid + s];
        __syncthreads();
    }
    if (tid == 0) partials[bid] = red[0];
}

// ---------------------- deterministic final reduce: acc[0]+=mean, acc[1]+=.25*mean
__global__ __launch_bounds__(256) void loss_reduce_kernel(const float* __restrict__ partials,
                                                          float* __restrict__ acc) {
    __shared__ float red[256];
    float s = 0.0f;
    for (int i = threadIdx.x; i < 1024; i += 256) s += partials[i];
    red[threadIdx.x] = s;
    __syncthreads();
    for (int st = 128; st > 0; st >>= 1) {
        if (threadIdx.x < st) red[threadIdx.x] += red[threadIdx.x + st];
        __syncthreads();
    }
    if (threadIdx.x == 0) {
        float mean = red[0] / (float)ELEMS;
        acc[0] += mean;           // vq: sum of per-scale means (divide by 7 later)
        acc[1] += 0.25f * mean;   // commit: BETA * mean
    }
}

// ------------------------------------------------------------- finalize
__global__ __launch_bounds__(256) void finalize_kernel(const float* __restrict__ f_hat,
                                                       const float* __restrict__ acc,
                                                       float* __restrict__ out) {
    int i = blockIdx.x * 256 + threadIdx.x;
    if (i < ELEMS) out[i] = f_hat[i];   // f_hat_out == f_hat numerically (STE)
    if (i == 0) { out[ELEMS] = acc[0] / 7.0f; out[ELEMS + 1] = acc[1]; }
}

// ================================================================ launcher
extern "C" void kernel_launch(void* const* d_in, const int* in_sizes, int n_in,
                              void* d_out, int out_size, void* d_ws, size_t ws_size,
                              hipStream_t stream) {
    const float* f     = (const float*)d_in[0];   // [8,32,64,64]
    const float* cb    = (const float*)d_in[1];   // [4096,32]
    const float* phi_w = (const float*)d_in[2];   // [4,32,32,3,3]
    const float* phi_b = (const float*)d_in[3];   // [4,32]
    float* out = (float*)d_out;

    char* ws = (char*)d_ws;
    float*    f_hat    = (float*)ws;     ws += (size_t)ELEMS * 4;
    float*    f_rest   = (float*)ws;     ws += (size_t)ELEMS * 4;
    float*    h_pre    = (float*)ws;     ws += (size_t)ELEMS * 4;
    float*    h_post   = (float*)ws;     ws += (size_t)ELEMS * 4;
    float*    rest_nc  = (float*)ws;     ws += (size_t)(32768 * CVAE) * 4;
    _Float16* cbn      = (_Float16*)ws;  ws += (size_t)(VOCAB * CVAE) * 2;
    int*      idxbuf   = (int*)ws;       ws += (size_t)32768 * 4;
    float*    partials = (float*)ws;     ws += 1024 * 4;
    float*    acc      = (float*)ws;     ws += 256;

    init_kernel<<<ELEMS / 256, 256, 0, stream>>>(f, f_hat, f_rest, acc);
    cbn_kernel<<<VOCAB / 256, 256, 0, stream>>>(cb, cbn);

    static const int PNS[7]     = {1, 2, 4, 8, 16, 32, 64};
    static const int PHI_IDX[7] = {0, 0, 1, 1, 2, 3, 3};

    // dynamic LDS: 256KB codebook + 16KB reduction scratch (WGP has 320KB)
    const size_t simsmem = (size_t)VOCAB * CVAE * 2 + 256 * 8 * (4 + 4);

    for (int si = 0; si < 7; ++si) {
        int pn = PNS[si];
        int N  = 8 * pn * pn;
        int k  = PHI_IDX[si];
        downsample_nc_kernel<<<(N * CVAE + 255) / 256, 256, 0, stream>>>(f_rest, rest_nc, pn);
        sim_argmax_kernel<<<(N + 127) / 128, 256, simsmem, stream>>>(rest_nc, cbn, idxbuf, N);
        gather_upsample_kernel<<<ELEMS / 256, 256, 0, stream>>>(idxbuf, cb, h_pre, pn);
        phi_conv_kernel<<<ELEMS / 256, 256, 0, stream>>>(h_pre,
                                                         phi_w + (size_t)k * CVAE * CVAE * 9,
                                                         phi_b + (size_t)k * CVAE, h_post);
        update_loss_kernel<<<1024, 256, 0, stream>>>(f, f_hat, f_rest, h_post, partials);
        loss_reduce_kernel<<<1, 256, 0, stream>>>(partials, acc);
    }

    finalize_kernel<<<(ELEMS + 2 + 255) / 256, 256, 0, stream>>>(f_hat, acc, out);
}
